// RetNet_34668976013941
// MI455X (gfx1250) — matrix operand order repaired and checked by reading, not compiled, not run on hardware
//
#include <hip/hip_runtime.h>
#include <cmath>

// ---------------------------------------------------------------------------
// CDNA5 (gfx1250) wave32 WMMA types & helpers
// ---------------------------------------------------------------------------
typedef __attribute__((ext_vector_type(16))) __bf16 v16bf;
typedef __attribute__((ext_vector_type(8)))  float  v8f;
typedef __attribute__((ext_vector_type(4)))  int    v4i;
typedef __attribute__((ext_vector_type(8)))  int    v8i;

__device__ __forceinline__ __bf16 cvt_bf16(float f) { return (__bf16)f; }

__device__ __forceinline__ v8f wmma_bf16(v16bf a, v16bf b, v8f c) {
  // D = A(16x32 bf16) * B(32x16 bf16) + C(16x16 f32)
  return __builtin_amdgcn_wmma_f32_16x16x32_bf16(
      /*neg_a=*/false, a, /*neg_b=*/false, b,
      /*c_mod=*/(short)0, c, /*reuse_a=*/false, /*reuse_b=*/false);
}

// ---------------------------------------------------------------------------
// Tensor Data Mover descriptor helpers (CDNA5 ISA)
// 2-D tile load: tile_dim0 = 32 f32 (contiguous), tile_dim1 = rows,
// tensor_dim0_stride = K.  LDS padding: +1 DWORD every 32 DWORDs -> row
// stride 33 floats (conflict-free LDS banking for fragment reads).
// ---------------------------------------------------------------------------
__device__ __forceinline__ v4i tdm_group0(const void* gptr, unsigned ldsoff) {
  unsigned long long ga = (unsigned long long)(uintptr_t)gptr;  // 57-bit VA
  v4i g;
  g[0] = 1;                                            // count=1, user mode
  g[1] = (int)ldsoff;                                  // lds_addr (bytes)
  g[2] = (int)(unsigned)(ga & 0xFFFFFFFFull);          // global_addr[31:0]
  g[3] = (int)((unsigned)((ga >> 32) & 0x01FFFFFFull)  // global_addr[56:32]
               | 0x80000000u);                         // type=2 ("image")
  return g;
}

__device__ __forceinline__ v8i tdm_group1(int K, int rows) {
  const int TD1 = 1 << 20;                 // tensor_dim1: large (no OOB use)
  v8i g;
  g[0] = 0x00020000                        // data_size = 4 bytes
       | (1 << 20)                         // pad_enable
       | (4 << 22);                        // pad_interval: 32 DWORDs
                                           // pad_amount field 0 -> 1 DWORD
  g[1] = (K & 0xFFFF) << 16;               // tensor_dim0[15:0] @ [63:48]
  g[2] = (int)(((unsigned)K >> 16) & 0xFFFFu) | ((TD1 & 0xFFFF) << 16);
  g[3] = (int)(((unsigned)TD1 >> 16) & 0xFFFFu) | (32 << 16);   // tile_dim0=32
  g[4] = rows;                             // tile_dim1 = rows, tile_dim2 = 0
  g[5] = K;                                // tensor_dim0_stride[31:0]
  g[6] = 0;                                // stride hi, dim1_stride lo
  g[7] = 0;
  return g;
}

__device__ __forceinline__ void tdm_load(v4i g0, v8i g1) {
  asm volatile("tensor_load_to_lds %0, %1" : : "s"(g0), "s"(g1) : "memory");
}

// ---------------------------------------------------------------------------
// Generic GEMM: out[M,N] = act(A[M,K] @ W[N,K]^T + bias) (+ res)
// Block tile 128x64, 8 waves (4x2), wave tile 32x32 (2x2 WMMA), K-step 32.
// Tiles staged fp32 in LDS by the TDM (double-buffered, async), fragments
// convert to bf16 with the native cvt on read.
// ---------------------------------------------------------------------------
#define BM 128
#define BN 64
#define BK 32

__global__ __launch_bounds__(256) void gemm_bias_kernel(
    const float* __restrict__ A, const float* __restrict__ W,
    const float* __restrict__ bias, const float* __restrict__ res,
    float* __restrict__ out, int M, int N, int K, int act)
{
  __shared__ float As[2][BM][BK + 1];   // +1 = TDM pad DWORD per 32
  __shared__ float Bs[2][BN][BK + 1];

  const int tid  = threadIdx.x;
  const int lane = tid & 31;
  const int wv   = tid >> 5;
  const int wm   = wv & 3;        // 4 waves along M
  const int wn   = wv >> 2;       // 2 waves along N
  const int lr   = lane & 15;
  const int hiq  = lane >> 4;
  const int kbA  = hiq * 16;      // A-frag k base
  const int kbB  = hiq * 16;      // B-frag k base
  const int r0   = blockIdx.y * BM;
  const int n0   = blockIdx.x * BN;

  const v8i g1A = tdm_group1(K, BM);
  const v8i g1B = tdm_group1(K, BN);

  v8f acc[2][2];
#pragma unroll
  for (int i = 0; i < 2; ++i)
#pragma unroll
    for (int j = 0; j < 2; ++j)
#pragma unroll
      for (int r = 0; r < 8; ++r) acc[i][j][r] = 0.f;

  const int nk = K / BK;

  // ---- preload tile 0 via TDM (wave 0 drives the DMA engine) ----
  if (wv == 0) {
    tdm_load(tdm_group0(A + (size_t)r0 * K, (unsigned)(uintptr_t)&As[0][0][0]), g1A);
    tdm_load(tdm_group0(W + (size_t)n0 * K, (unsigned)(uintptr_t)&Bs[0][0][0]), g1B);
    __builtin_amdgcn_s_wait_tensorcnt(0);
  }
  __syncthreads();

  for (int t = 0; t < nk; ++t) {
    const int cur = t & 1;
    // kick off DMA for tile t+1 into the other buffer (overlaps compute)
    if (wv == 0 && t + 1 < nk) {
      const int nxt = cur ^ 1;
      const size_t ko = (size_t)(t + 1) * BK;
      tdm_load(tdm_group0(A + (size_t)r0 * K + ko,
                          (unsigned)(uintptr_t)&As[nxt][0][0]), g1A);
      tdm_load(tdm_group0(W + (size_t)n0 * K + ko,
                          (unsigned)(uintptr_t)&Bs[nxt][0][0]), g1B);
    }

    // A fragment: lane = M row, k = hiq*16 + e
    v16bf af[2], bfr[2];
#pragma unroll
    for (int i = 0; i < 2; ++i) {
      const float* Ar = As[cur][wm * 32 + i * 16 + lr];
#pragma unroll
      for (int e = 0; e < 8; ++e) {
        af[i][e]     = cvt_bf16(Ar[kbA + e]);
        af[i][e + 8] = cvt_bf16(Ar[kbA + 8 + e]);
      }
    }
    // B fragment: lane = N col, k = hiq*16 + e  (B[k][n] = W[n][k])
#pragma unroll
    for (int j = 0; j < 2; ++j) {
      const float* Br = Bs[cur][wn * 32 + j * 16 + lr];
#pragma unroll
      for (int e = 0; e < 16; ++e) bfr[j][e] = cvt_bf16(Br[kbB + e]);
    }
#pragma unroll
    for (int i = 0; i < 2; ++i)
#pragma unroll
      for (int j = 0; j < 2; ++j)
        acc[i][j] = wmma_bf16(af[i], bfr[j], acc[i][j]);

    if (wv == 0 && t + 1 < nk) __builtin_amdgcn_s_wait_tensorcnt(0);
    __syncthreads();
  }

  // epilogue: C/D layout -> lane = N col, VGPR r -> M = r + 8*hiq
#pragma unroll
  for (int i = 0; i < 2; ++i) {
#pragma unroll
    for (int j = 0; j < 2; ++j) {
      int col = n0 + wn * 32 + j * 16 + lr;
      float bia = bias ? bias[col] : 0.f;
#pragma unroll
      for (int r = 0; r < 8; ++r) {
        int row = r0 + wm * 32 + i * 16 + hiq * 8 + r;
        float val = acc[i][j][r] + bia;
        if (act == 1) val = val / (1.f + __expf(-val));
        if (res) val += res[(size_t)row * N + col];
        out[(size_t)row * N + col] = val;
      }
    }
  }
}

// ---------------------------------------------------------------------------
// LayerNorm over rows of length N (no affine)
// ---------------------------------------------------------------------------
__global__ __launch_bounds__(256) void layernorm_kernel(
    const float* __restrict__ in, float* __restrict__ out, int N)
{
  __shared__ float red[256];
  const size_t base = (size_t)blockIdx.x * N;
  float s = 0.f;
  for (int i = threadIdx.x; i < N; i += 256) s += in[base + i];
  red[threadIdx.x] = s; __syncthreads();
  for (int off = 128; off > 0; off >>= 1) {
    if (threadIdx.x < off) red[threadIdx.x] += red[threadIdx.x + off];
    __syncthreads();
  }
  float mean = red[0] / N;
  __syncthreads();
  float vv = 0.f;
  for (int i = threadIdx.x; i < N; i += 256) {
    float d = in[base + i] - mean; vv += d * d;
  }
  red[threadIdx.x] = vv; __syncthreads();
  for (int off = 128; off > 0; off >>= 1) {
    if (threadIdx.x < off) red[threadIdx.x] += red[threadIdx.x + off];
    __syncthreads();
  }
  float rstd = rsqrtf(red[0] / N + 1e-5f);
  for (int i = threadIdx.x; i < N; i += 256)
    out[base + i] = (in[base + i] - mean) * rstd;
}

// ---------------------------------------------------------------------------
// Per-layer phase tables: cos(l*th), sin(l*th), a^(l+1)*cos((l+1)th), ... sin
// idx -> [l, h, d] over L*H*DQ = 1024*8*128
// ---------------------------------------------------------------------------
__global__ __launch_bounds__(256) void tables_kernel(
    const float* __restrict__ pre, const float* __restrict__ pim,
    const float* __restrict__ amp,
    float* __restrict__ cost, float* __restrict__ sint,
    float* __restrict__ wrt, float* __restrict__ wit)
{
  int idx = blockIdx.x * 256 + threadIdx.x;
  int d  = idx & 127;
  int hh = (idx >> 7) & 7;
  int l  = idx >> 10;
  float re = pre[hh * 128 + d], im = pim[hh * 128 + d];
  float th = atan2f(im, re);                          // phase = e^{i th}
  float sn, cs;
  sincosf((float)l * th, &sn, &cs);
  cost[idx] = cs; sint[idx] = sn;
  float a = 1.f / (1.f + __expf(-amp[hh]));           // sigmoid
  float mag = __expf((float)(l + 1) * __logf(a));     // a^(l+1)
  float sn2, cs2;
  sincosf((float)(l + 1) * th, &sn2, &cs2);
  wrt[idx] = mag * cs2; wit[idx] = mag * sn2;         // phazor^(l+1)
}

// ---------------------------------------------------------------------------
// Fused causal retention. One wave per (b, h, 16-row l-tile).
//   O(16x128) = cross(q*w @ lc) ; then flash loop over 32-wide m blocks:
//   S(16x32) = qR@kR^T + qI@kI^T ; S *= a^(l-m) (l>=m) ; O += S@V
//   y = silu(O)
// ---------------------------------------------------------------------------
#define L_  1024
#define H_  8
#define DQ_ 128
#define HD_ 1024

__global__ __launch_bounds__(128) void retention_kernel(
    const float* __restrict__ q, const float* __restrict__ k,
    const float* __restrict__ v,
    const float* __restrict__ cost, const float* __restrict__ sint,
    const float* __restrict__ wrt,  const float* __restrict__ wit,
    const float* __restrict__ lcr,  const float* __restrict__ lci,
    const float* __restrict__ amp,  float* __restrict__ y)
{
  __shared__ __bf16 sS[4][16][34];           // per-wave S staging

  const int tid  = threadIdx.x;
  const int lane = tid & 31;
  const int w    = tid >> 5;
  const int lr   = lane & 15;
  const int hiq  = lane >> 4;
  const int gw   = blockIdx.x * 4 + w;       // 0..1023 = B*H*(L/16)
  const int lt   = gw & 63;
  const int hh   = (gw >> 6) & 7;
  const int b    = gw >> 9;
  const int l0   = lt * 16;
  const int kbA  = hiq * 16;                 // A-frag k base
  const int kbB  = hiq * 16;                 // B-frag k base

  const float a_h  = 1.f / (1.f + __expf(-amp[hh]));
  const float ln_a = __logf(a_h);

  const int lrow = l0 + lr;
  const float* qrow = q + ((size_t)(b * L_ + lrow)) * HD_ + hh * DQ_;
  const size_t tabrow = ((size_t)lrow * H_ + hh) * DQ_;
  const float* lcrh = lcr + (size_t)hh * DQ_ * DQ_;
  const float* lcih = lci + (size_t)hh * DQ_ * DQ_;

  v8f O[8];
#pragma unroll
  for (int j = 0; j < 8; ++j)
#pragma unroll
    for (int r = 0; r < 8; ++r) O[j][r] = 0.f;

  // ---- cross term: O = (q*wr) @ lc_re + (-(q*wi)) @ lc_im  (real part) ----
#pragma unroll
  for (int kk = 0; kk < 4; ++kk) {
    v16bf aR, aI;
#pragma unroll
    for (int e = 0; e < 8; ++e) {
      int d1 = kk * 32 + kbA + e;
      int d2 = d1 + 8;
      float q1 = qrow[d1], q2 = qrow[d2];
      aR[e]     = cvt_bf16(q1 * wrt[tabrow + d1]);
      aR[e + 8] = cvt_bf16(q2 * wrt[tabrow + d2]);
      aI[e]     = cvt_bf16(-q1 * wit[tabrow + d1]);
      aI[e + 8] = cvt_bf16(-q2 * wit[tabrow + d2]);
    }
#pragma unroll
    for (int j = 0; j < 8; ++j) {
      int n = j * 16 + lr;                     // output DQ column
      v16bf bR, bI;
#pragma unroll
      for (int e = 0; e < 16; ++e) {
        int dd = kk * 32 + kbB + e;            // contraction index
        bR[e] = cvt_bf16(lcrh[(size_t)dd * DQ_ + n]);
        bI[e] = cvt_bf16(lcih[(size_t)dd * DQ_ + n]);
      }
      O[j] = wmma_bf16(aR, bR, O[j]);
      O[j] = wmma_bf16(aI, bI, O[j]);
    }
  }

  // ---- hoisted query fragments qR = q*cos(l th), qI = q*sin(l th) ----
  v16bf qRf[4], qIf[4];
#pragma unroll
  for (int kk = 0; kk < 4; ++kk) {
#pragma unroll
    for (int e = 0; e < 8; ++e) {
      int d1 = kk * 32 + kbA + e;
      int d2 = d1 + 8;
      float q1 = qrow[d1], q2 = qrow[d2];
      qRf[kk][e]     = cvt_bf16(q1 * cost[tabrow + d1]);
      qRf[kk][e + 8] = cvt_bf16(q2 * cost[tabrow + d2]);
      qIf[kk][e]     = cvt_bf16(q1 * sint[tabrow + d1]);
      qIf[kk][e + 8] = cvt_bf16(q2 * sint[tabrow + d2]);
    }
  }

  // ---- causal flash loop over 32-wide key blocks ----
  const int nmb = (l0 + 47) >> 5;
  for (int mb = 0; mb < nmb; ++mb) {
    const int m0 = mb * 32;
    v8f s0, s1;
#pragma unroll
    for (int r = 0; r < 8; ++r) { s0[r] = 0.f; s1[r] = 0.f; }

#pragma unroll
    for (int kk = 0; kk < 4; ++kk) {
#pragma unroll
      for (int j = 0; j < 2; ++j) {
        int m = m0 + j * 16 + lr;
        const float* krow = k + ((size_t)(b * L_ + m)) * HD_ + hh * DQ_;
        const size_t trow = ((size_t)m * H_ + hh) * DQ_;
        v16bf kR, kI;
#pragma unroll
        for (int e = 0; e < 16; ++e) {
          int dd = kk * 32 + kbB + e;
          float kvv = krow[dd];
          kR[e] = cvt_bf16(kvv * cost[trow + dd]);
          kI[e] = cvt_bf16(kvv * sint[trow + dd]);
        }
        if (j == 0) { s0 = wmma_bf16(qRf[kk], kR, s0); s0 = wmma_bf16(qIf[kk], kI, s0); }
        else        { s1 = wmma_bf16(qRf[kk], kR, s1); s1 = wmma_bf16(qIf[kk], kI, s1); }
      }
    }

    // decay mask a^(l-m) (causal), stage S as bf16 in LDS (C-layout -> tile)
#pragma unroll
    for (int r = 0; r < 8; ++r) {
      int ll  = l0 + hiq * 8 + r;
      int m_a = m0 + lr;
      int m_b = m0 + 16 + lr;
      float sv0 = (ll >= m_a) ? s0[r] * __expf((float)(ll - m_a) * ln_a) : 0.f;
      float sv1 = (ll >= m_b) ? s1[r] * __expf((float)(ll - m_b) * ln_a) : 0.f;
      sS[w][hiq * 8 + r][lr]      = cvt_bf16(sv0);
      sS[w][hiq * 8 + r][16 + lr] = cvt_bf16(sv1);
    }
    __builtin_amdgcn_wave_barrier();
    asm volatile("s_wait_dscnt 0" ::: "memory");

    // re-fragment S (16x32) as WMMA A operand
    v16bf sa;
#pragma unroll
    for (int e = 0; e < 8; ++e) {
      sa[e]     = sS[w][lr][kbA + e];
      sa[e + 8] = sS[w][lr][kbA + 8 + e];
    }
    // O += S @ V ; B[k=m][n=d] = v[b, m, h, d]
#pragma unroll
    for (int j2 = 0; j2 < 8; ++j2) {
      int dd = j2 * 16 + lr;
      v16bf vf;
#pragma unroll
      for (int e = 0; e < 16; ++e) {
        int m = m0 + kbB + e;
        vf[e] = cvt_bf16(v[((size_t)(b * L_ + m)) * HD_ + hh * DQ_ + dd]);
      }
      O[j2] = wmma_bf16(sa, vf, O[j2]);
    }
    __builtin_amdgcn_wave_barrier();
  }

  // ---- epilogue: y = silu(O) ----
#pragma unroll
  for (int j2 = 0; j2 < 8; ++j2) {
    int dd = j2 * 16 + lr;
#pragma unroll
    for (int r = 0; r < 8; ++r) {
      int ll = l0 + hiq * 8 + r;
      float val = O[j2][r];
      val = val / (1.f + __expf(-val));
      y[((size_t)(b * L_ + ll)) * HD_ + hh * DQ_ + dd] = val;
    }
  }
}

// ---------------------------------------------------------------------------
// Host orchestration
// ---------------------------------------------------------------------------
extern "C" void kernel_launch(void* const* d_in, const int* in_sizes, int n_in,
                              void* d_out, int out_size, void* d_ws, size_t ws_size,
                              hipStream_t stream) {
  (void)in_sizes; (void)n_in; (void)out_size; (void)ws_size;

  const float* x      = (const float*)d_in[0];
  const float* tin_w  = (const float*)d_in[1];
  const float* tin_b  = (const float*)d_in[2];
  const float* tout_w = (const float*)d_in[3];
  const float* tout_b = (const float*)d_in[4];
  const float* wq_w   = (const float*)d_in[5];
  const float* wq_b   = (const float*)d_in[6];
  const float* wk_w   = (const float*)d_in[7];
  const float* wk_b   = (const float*)d_in[8];
  const float* wv_w   = (const float*)d_in[9];
  const float* wv_b   = (const float*)d_in[10];
  const float* wo_w   = (const float*)d_in[11];
  const float* wo_b   = (const float*)d_in[12];
  const float* f1_w   = (const float*)d_in[13];
  const float* f1_b   = (const float*)d_in[14];
  const float* f2_w   = (const float*)d_in[15];
  const float* f2_b   = (const float*)d_in[16];
  const float* phz_re = (const float*)d_in[17];
  const float* phz_im = (const float*)d_in[18];
  const float* amp    = (const float*)d_in[19];
  const float* lc_re  = (const float*)d_in[20];
  const float* lc_im  = (const float*)d_in[21];

  const int M = 2048, DIMc = 1024, DHc = 4096, Vc = 32000;
  const size_t SM  = (size_t)M * DIMc;   // 2M floats
  const size_t TSZ = (size_t)1024 * 8 * 128;

  float* ws   = (float*)d_ws;
  float* h    = ws;
  float* ln   = h   + SM;
  float* qb   = ln  + SM;
  float* kb   = qb  + SM;
  float* vb   = kb  + SM;
  float* yb   = vb  + SM;
  float* ffn  = yb  + SM;
  float* cost = ffn + (size_t)M * DHc;
  float* sint = cost + TSZ;
  float* wrt  = sint + TSZ;
  float* wit  = wrt  + TSZ;

  auto gemm = [&](const float* A, const float* W, const float* bias,
                  const float* res, float* out, int MM, int NN, int KK, int act) {
    dim3 grid(NN / BN, MM / BM);
    gemm_bias_kernel<<<grid, 256, 0, stream>>>(A, W, bias, res, out, MM, NN, KK, act);
  };

  // token-in projection
  gemm(x, tin_w, tin_b, nullptr, h, M, DIMc, Vc, 0);

  for (int i = 0; i < 4; ++i) {
    layernorm_kernel<<<M, 256, 0, stream>>>(h, ln, DIMc);
    tables_kernel<<<(1024 * 8 * 128) / 256, 256, 0, stream>>>(
        phz_re + (size_t)i * 8 * 128, phz_im + (size_t)i * 8 * 128,
        amp + (size_t)i * 8, cost, sint, wrt, wit);

    gemm(ln, wq_w + (size_t)i * DIMc * DIMc, wq_b + (size_t)i * DIMc, nullptr, qb, M, DIMc, DIMc, 0);
    gemm(ln, wk_w + (size_t)i * DIMc * DIMc, wk_b + (size_t)i * DIMc, nullptr, kb, M, DIMc, DIMc, 0);
    gemm(ln, wv_w + (size_t)i * DIMc * DIMc, wv_b + (size_t)i * DIMc, nullptr, vb, M, DIMc, DIMc, 0);

    retention_kernel<<<256, 128, 0, stream>>>(
        qb, kb, vb, cost, sint, wrt, wit,
        lc_re + (size_t)i * 8 * 128 * 128, lc_im + (size_t)i * 8 * 128 * 128,
        amp + (size_t)i * 8, yb);

    gemm(yb, wo_w + (size_t)i * DIMc * DIMc, wo_b + (size_t)i * DIMc, h, h, M, DIMc, DIMc, 0);

    layernorm_kernel<<<M, 256, 0, stream>>>(h, ln, DIMc);
    gemm(ln,  f1_w + (size_t)i * DHc * DIMc, f1_b + (size_t)i * DHc, nullptr, ffn, M, DHc, DIMc, 1);
    gemm(ffn, f2_w + (size_t)i * DIMc * DHc, f2_b + (size_t)i * DIMc, h, h, M, DIMc, DHc, 0);
  }

  // token-out projection
  gemm(h, tout_w, tout_b, nullptr, (float*)d_out, M, Vc, DIMc, 0);
}
